// SAGNet_HIER_11501922419053
// MI455X (gfx1250) — compile-verified, matrix operand order checked by reading, not run on hardware
//
#include <hip/hip_runtime.h>
#include <hip/hip_bf16.h>
#include <stdint.h>

// Problem constants (match reference)
#define NN   131072
#define EE   1048576
#define BB   512
#define FF   78
#define TL   1000
#define NVOC 26
#define EMBD 128
#define HPAD 96   // padded lda for the 78-wide node features (multiple of 32)

typedef __attribute__((ext_vector_type(16))) _Float16 v16h;
typedef __attribute__((ext_vector_type(8)))  _Float16 v8h;
typedef __attribute__((ext_vector_type(8)))  float    v8f;

// ---------------------------------------------------------------------------
// Fragment-native packing (once per launch, cold path, bounds checks OK)
//
// WMMA 16x16x32 f16 fragment, wave32:
//   lane (l<16): rows/cols = l&15, kh = (lane>>4)*8
//   flattened element e (0..15): k = kh + ((e&8)?16:0) + (e&7)
//
// Packed layout: tile-major, per-lane 16 halves contiguous (32B):
//   B-pack: off = ((nt*Kt + kt)*32 + lane)*16   (kt-contiguous streaming)
//   A-pack: off = ((mt*Kt + kt)*32 + lane)*16
// ---------------------------------------------------------------------------
__global__ void k_pack_b(const float* __restrict__ src, _Float16* __restrict__ dst,
                         int K, int N, int Kt, int Nt) {
  const int tid = blockIdx.x * blockDim.x + threadIdx.x;
  const int lane = tid & 31;
  const int kt = (tid >> 5) % Kt;
  const int nt = (tid >> 5) / Kt;
  if (nt >= Nt) return;
  const int n = nt * 16 + (lane & 15);
  const int kh = (lane >> 4) * 8;
  _Float16* out = dst + ((size_t)(nt * Kt + kt) * 32 + lane) * 16;
#pragma unroll
  for (int e = 0; e < 16; ++e) {
    const int k = kt * 32 + kh + ((e & 8) ? 16 : 0) + (e & 7);
    const float v = (k < K && n < N) ? src[(size_t)k * N + n] : 0.f;
    out[e] = (_Float16)v;
  }
}

__global__ void k_pack_a(const float* __restrict__ src, _Float16* __restrict__ dst,
                         int M, int K, int Kt) {
  const int tid = blockIdx.x * blockDim.x + threadIdx.x;
  const int lane = tid & 31;
  const int kt = (tid >> 5) % Kt;
  const int mt = (tid >> 5) / Kt;
  if (mt >= M / 16) return;
  const int m = mt * 16 + (lane & 15);
  const int kh = (lane >> 4) * 8;
  _Float16* out = dst + ((size_t)(mt * Kt + kt) * 32 + lane) * 16;
#pragma unroll
  for (int e = 0; e < 16; ++e) {
    const int k = kt * 32 + kh + ((e & 8) ? 16 : 0) + (e & 7);
    const float v = (k < K) ? src[(size_t)m * K + k] : 0.f;
    out[e] = (_Float16)v;
  }
}

// f32 [rows, cols] -> f16 [rows, dstride] with zeroed pad columns
__global__ void k_cvt_pad(const float* __restrict__ s, _Float16* __restrict__ d,
                          int rows, int cols, int dstride) {
  const int total = rows * dstride;
  for (int i = blockIdx.x * blockDim.x + threadIdx.x; i < total;
       i += gridDim.x * blockDim.x) {
    const int r = i / dstride;
    const int c = i - r * dstride;
    d[i] = (c < cols) ? (_Float16)s[(size_t)r * cols + c] : (_Float16)0.f;
  }
}

__global__ void k_f32_to_f16(const float* __restrict__ s, _Float16* __restrict__ d,
                             int n) {
  for (int i = blockIdx.x * blockDim.x + threadIdx.x; i < n;
       i += gridDim.x * blockDim.x)
    d[i] = (_Float16)s[i];
}

__device__ __forceinline__ v16h frag_a(const _Float16* __restrict__ p) {
  const v8h lo = *(const v8h*)p;
  const v8h hi = *(const v8h*)(p + 16);
  return __builtin_shufflevector(lo, hi, 0, 1, 2, 3, 4, 5, 6, 7, 8, 9, 10, 11, 12,
                                 13, 14, 15);
}

// ---------------------------------------------------------------------------
// Generic WMMA GEMM: C[M,N] = A[M,K] @ B[K,N], MT m-tiles per wave
//   A: f16 row-major, lda multiple of 32 halves, zero-padded; M mult of 16*MT
//   B: fragment-packed (k_pack_b)
// MODE 0: raw f32 store (ldc = N)
// MODE 1: (c+bias)*bnScale+bnShift, relu -> f16 at [m, coff+n]
// MODE 2: relu(c+bias), then bn          -> f16 at [m, coff+n]
// ---------------------------------------------------------------------------
template <int MODE, int MT>
__global__ void k_wmma_gemm(const _Float16* __restrict__ A, int lda,
                            const _Float16* __restrict__ Bp,
                            const float* __restrict__ bias,
                            const float* __restrict__ bn,
                            float* __restrict__ Cf, _Float16* __restrict__ Ch,
                            int M, int N, int Kt, int Nt, int ldc, int coff) {
  const int nwaves = blockDim.x >> 5;
  const int wave   = threadIdx.x >> 5;
  const int lane   = threadIdx.x & 31;
  const int m0     = blockIdx.x * 16 * MT;
  const int ntile  = blockIdx.y * nwaves + wave;
  if (ntile >= Nt) return;

  const _Float16* arow[MT];
#pragma unroll
  for (int t = 0; t < MT; ++t)
    arow[t] = A + (size_t)(m0 + t * 16 + (lane & 15)) * lda + ((lane >> 4) * 8);
  const _Float16* bptr = Bp + ((size_t)ntile * Kt * 32 + lane) * 16;

  const v8f zero = {};
  v8f c[MT];
#pragma unroll
  for (int t = 0; t < MT; ++t) c[t] = zero;

  for (int kt = 0; kt < Kt; ++kt) {
    __builtin_prefetch(bptr + 512 * 4, 0, 1);
    const v16h b = *(const v16h*)bptr;
    bptr += 512;
#pragma unroll
    for (int t = 0; t < MT; ++t) {
      const v16h a = frag_a(arow[t]);
      c[t] = __builtin_amdgcn_wmma_f32_16x16x32_f16(false, a, false, b, (short)0,
                                                    c[t], false, false);
      arow[t] += 32;
    }
  }

  const int n  = ntile * 16 + (lane & 15);
  const int mb = (lane >> 4) * 8;
  const float rs = rsqrtf(1.0f + 1e-5f);
  float bi = 0.f, sc = 0.f, sh = 0.f;
  if (MODE != 0 && n < N) { bi = bias[n]; sc = bn[n] * rs; sh = bn[N + n]; }
  if (n < N) {
#pragma unroll
    for (int t = 0; t < MT; ++t) {
#pragma unroll
      for (int r = 0; r < 8; ++r) {
        const int m = m0 + t * 16 + mb + r;
        float v = c[t][r];
        if (MODE == 0) {
          Cf[(size_t)m * ldc + n] = v;
        } else if (MODE == 1) {
          v = (v + bi) * sc + sh;
          v = fmaxf(v, 0.f);
          Ch[(size_t)m * ldc + coff + n] = (_Float16)v;
        } else {
          v = fmaxf(v + bi, 0.f);
          v = v * sc + sh;
          Ch[(size_t)m * ldc + coff + n] = (_Float16)v;
        }
      }
    }
  }
}

// ---------------------------------------------------------------------------
// Conv1d(kernel=3, pad=1) as implicit-im2col WMMA GEMM.
//   Out[b][m][l] = relu(bn(sum_{k=i*3+t} W[m][k] * X[b][i][l+t-1] + bias[m]))
// One block per batch element. 8 waves = 8 l-tiles. Each wave builds its
// im2col B fragment ONCE per k-step (16 LDS gathers) and feeds all MT m-tile
// accumulators -> MT independent WMMA chains, MT x fewer LDS gathers.
//   EMBED: X[i][.] = emb[target[b][i]][.], emb table staged in LDS.
// ---------------------------------------------------------------------------
template <bool EMBED, int CIN, int MT>
__global__ void k_wmma_conv(const _Float16* __restrict__ Ap,
                            const _Float16* __restrict__ Xin,
                            const _Float16* __restrict__ embF,
                            const int* __restrict__ target,
                            const float* __restrict__ bias,
                            const float* __restrict__ bn,
                            _Float16* __restrict__ Out,
                            int M, int Kt, int Kreal) {
  const int b = blockIdx.x;
  constexpr int XS = EMBED ? (NVOC * EMBD + 8) : (CIN * 128 + 8);
  __shared__ __align__(16) _Float16 Xs[XS];
  __shared__ int tgtS[EMBED ? 1008 : 8];

  if (EMBED) {
    for (int i = threadIdx.x; i < 1008; i += blockDim.x)
      tgtS[i] = (i < TL) ? target[b * TL + i] : 0;
    for (int i = threadIdx.x; i < XS; i += blockDim.x)
      Xs[i] = (i < NVOC * EMBD) ? embF[i] : (_Float16)0.f;
  } else {
    const uint32_t* s32 = (const uint32_t*)(Xin + (size_t)b * CIN * 128);
    uint32_t* d32 = (uint32_t*)Xs;
    for (int i = threadIdx.x; i < CIN * 64; i += blockDim.x) d32[i] = s32[i];
    if (threadIdx.x < 8) Xs[CIN * 128 + threadIdx.x] = (_Float16)0.f;
  }
  __syncthreads();
  const int ZS = EMBED ? (NVOC * EMBD) : (CIN * 128);

  const int wave = threadIdx.x >> 5;
  const int lane = threadIdx.x & 31;
  const int l    = wave * 16 + (lane & 15);
  const int kh   = (lane >> 4) * 8;

  const _Float16* aptr[MT];
#pragma unroll
  for (int t = 0; t < MT; ++t)
    aptr[t] = Ap + ((size_t)(t * Kt) * 32 + lane) * 16;

  const v8f zero = {};
  v8f c[MT];
#pragma unroll
  for (int t = 0; t < MT; ++t) c[t] = zero;

  for (int kt = 0; kt < Kt; ++kt) {
    const int kb = kt * 32 + kh;
    v16h bb;
#pragma unroll
    for (int e = 0; e < 16; ++e) {
      const int kk = kb + ((e & 8) ? 16 : 0) + (e & 7);
      const int i  = kk / 3;
      const int t  = kk - 3 * i;
      const int lp = l + t - 1;
      const bool valid = ((unsigned)lp < 128u) && (kk < Kreal);
      const int base = EMBED ? (tgtS[i] * 128) : (i * 128);
      bb[e] = Xs[valid ? (base + lp) : ZS];
    }
#pragma unroll
    for (int t = 0; t < MT; ++t) {
      __builtin_prefetch(aptr[t] + 512 * 2, 0, 1);
      const v16h a = *(const v16h*)aptr[t];
      c[t] = __builtin_amdgcn_wmma_f32_16x16x32_f16(false, a, false, bb, (short)0,
                                                    c[t], false, false);
      aptr[t] += 512;
    }
  }

  const int mb = (lane >> 4) * 8;
  const float rs = rsqrtf(1.0f + 1e-5f);
#pragma unroll
  for (int t = 0; t < MT; ++t) {
#pragma unroll
    for (int r = 0; r < 8; ++r) {
      const int m = t * 16 + mb + r;
      float v = c[t][r] + bias[m];
      v = v * (bn[m] * rs) + bn[M + m];
      v = fmaxf(v, 0.f);
      Out[((size_t)b * M + m) * 128 + l] = (_Float16)v;
    }
  }
}

// ---------------------------------------------------------------------------
// Graph / elementwise kernels (fp32, memory-bound)
// ---------------------------------------------------------------------------
__global__ void k_deg_count(const int* __restrict__ dst, float* __restrict__ deg) {
  const int e = blockIdx.x * blockDim.x + threadIdx.x;
  if (e < EE) atomicAdd(&deg[dst[e]], 1.0f);
}

__global__ void k_deg_finish(float* __restrict__ dis) {
  const int i = blockIdx.x * blockDim.x + threadIdx.x;
  if (i < NN) dis[i] = rsqrtf(dis[i] + 1.0f);
}

__global__ void k_edge_scatter(const float* __restrict__ hw,
                               const int* __restrict__ src,
                               const int* __restrict__ dst,
                               const float* __restrict__ dis,
                               float* __restrict__ agg) {
  const int tid = blockIdx.x * blockDim.x + threadIdx.x;
  const int e = tid >> 7, f = tid & 127;
  if (e >= EE || f >= FF) return;
  const int s = src[e], d = dst[e];
  const float coef = dis[s] * dis[d];
  atomicAdd(&agg[(size_t)d * FF + f], hw[(size_t)s * FF + f] * coef);
}

__global__ void k_combine(const float* __restrict__ agg, const float* __restrict__ hw,
                          const float* __restrict__ dis, const float* __restrict__ bias,
                          const float* __restrict__ bn, float* __restrict__ h) {
  const int tid = blockIdx.x * blockDim.x + threadIdx.x;
  const int i = tid >> 7, f = tid & 127;
  if (i >= NN || f >= FF) return;
  const float rs = rsqrtf(1.0f + 1e-5f);
  const float d = dis[i];
  const size_t o = (size_t)i * FF + f;
  float v = agg[o] + hw[o] * d * d + bias[f];
  v = v * (bn[f] * rs) + bn[FF + f];
  h[o] = fmaxf(v, 0.f);
}

__global__ void k_pool_dot(const float* __restrict__ h, const float* __restrict__ Wp,
                           float* __restrict__ hwp) {
  const int i = blockIdx.x * blockDim.x + threadIdx.x;
  if (i >= NN) return;
  const float* row = h + (size_t)i * FF;
  float acc = 0.f;
  for (int f = 0; f < FF; ++f) acc += row[f] * Wp[f];
  hwp[i] = acc;
}

__global__ void k_edge_scalar(const float* __restrict__ hwp,
                              const int* __restrict__ src,
                              const int* __restrict__ dst,
                              const float* __restrict__ dis,
                              float* __restrict__ aggs) {
  const int e = blockIdx.x * blockDim.x + threadIdx.x;
  if (e >= EE) return;
  const int s = src[e], d = dst[e];
  atomicAdd(&aggs[d], hwp[s] * dis[s] * dis[d]);
}

// h *= tanh(s) in place; also writes padded f16 copy for the next WMMA GEMM
__global__ void k_finalize(float* __restrict__ h, _Float16* __restrict__ hf,
                           const float* __restrict__ aggs,
                           const float* __restrict__ hwp,
                           const float* __restrict__ dis,
                           const float* __restrict__ bp) {
  const int tid = blockIdx.x * blockDim.x + threadIdx.x;
  const int i = tid >> 7, f = tid & 127;
  if (i >= NN || f >= FF) return;
  const float d = dis[i];
  const float s = aggs[i] + hwp[i] * d * d + bp[0];
  const float th = tanhf(s);
  const size_t o = (size_t)i * FF + f;
  const float v = h[o] * th;
  h[o] = v;
  hf[(size_t)i * HPAD + f] = (_Float16)v;
}

// readout per graph: 256 consecutive nodes (batch = repeat(arange(B), 256))
__global__ void k_readout(const float* __restrict__ h, float* __restrict__ xsum,
                          int first) {
  const int b = blockIdx.x;
  const int f = threadIdx.x;
  if (f >= FF) return;
  const float* base = h + ((size_t)b * 256) * FF + f;
  float mx = -3.402823466e38f, sm = 0.f;
  for (int r = 0; r < 256; ++r) {
    const float v = base[(size_t)r * FF];
    mx = fmaxf(mx, v);
    sm += v;
  }
  const float mean = sm * (1.0f / 256.0f);
  if (first) {
    xsum[b * 156 + f]      = mx;
    xsum[b * 156 + 78 + f] = mean;
  } else {
    xsum[b * 156 + f]      += mx;
    xsum[b * 156 + 78 + f] += mean;
  }
}

__global__ void k_out_dot(const _Float16* __restrict__ xc,
                          const float* __restrict__ w,
                          const float* __restrict__ ob, float* __restrict__ out) {
  const int b = blockIdx.x * blockDim.x + threadIdx.x;
  if (b >= BB) return;
  float acc = ob[0];
  const _Float16* row = xc + (size_t)b * 512;
  for (int k = 0; k < 512; ++k) acc += (float)row[k] * w[k];
  out[b] = acc;
}

// ---------------------------------------------------------------------------
extern "C" void kernel_launch(void* const* d_in, const int* in_sizes, int n_in,
                              void* d_out, int out_size, void* d_ws, size_t ws_size,
                              hipStream_t stream) {
  (void)in_sizes; (void)n_in; (void)out_size; (void)ws_size;

  const float* x      = (const float*)d_in[0];
  const int*   ei     = (const int*)d_in[1];
  const int*   src    = ei;
  const int*   dst    = ei + EE;
  const int*   target = (const int*)d_in[3];
  const float* W[3]   = {(const float*)d_in[4],  (const float*)d_in[9],  (const float*)d_in[14]};
  const float* bv[3]  = {(const float*)d_in[5],  (const float*)d_in[10], (const float*)d_in[15]};
  const float* bnv[3] = {(const float*)d_in[6],  (const float*)d_in[11], (const float*)d_in[16]};
  const float* Wp[3]  = {(const float*)d_in[7],  (const float*)d_in[12], (const float*)d_in[17]};
  const float* bp[3]  = {(const float*)d_in[8],  (const float*)d_in[13], (const float*)d_in[18]};
  const float* fcg1W  = (const float*)d_in[19];
  const float* fcg1b  = (const float*)d_in[20];
  const float* bn4    = (const float*)d_in[21];
  const float* emb    = (const float*)d_in[22];
  const float* c1W    = (const float*)d_in[23];
  const float* c1b    = (const float*)d_in[24];
  const float* bnx1   = (const float*)d_in[25];
  const float* c2W    = (const float*)d_in[26];
  const float* c2b    = (const float*)d_in[27];
  const float* bnx2   = (const float*)d_in[28];
  const float* c3W    = (const float*)d_in[29];
  const float* c3b    = (const float*)d_in[30];
  const float* bnx3   = (const float*)d_in[31];
  const float* fxtW   = (const float*)d_in[32];
  const float* fxtb   = (const float*)d_in[33];
  const float* bn6    = (const float*)d_in[34];
  const float* fc1W   = (const float*)d_in[35];
  const float* fc1b   = (const float*)d_in[36];
  const float* bn7    = (const float*)d_in[37];
  const float* fc2W   = (const float*)d_in[38];
  const float* fc2b   = (const float*)d_in[39];
  const float* bn8    = (const float*)d_in[40];
  const float* outW   = (const float*)d_in[41];
  const float* outb   = (const float*)d_in[42];
  float* out = (float*)d_out;

  // bump allocator in workspace
  uintptr_t cur = (uintptr_t)d_ws;
  auto alloc = [&](size_t bytes) -> void* {
    cur = (cur + 255) & ~(uintptr_t)255;
    void* r = (void*)cur;
    cur += bytes;
    return r;
  };
  _Float16* hf16  = (_Float16*)alloc((size_t)NN * HPAD * 2);
  float*    hw    = (float*)alloc((size_t)NN * FF * 4);
  float*    agg   = (float*)alloc((size_t)NN * FF * 4);
  float*    h     = (float*)alloc((size_t)NN * FF * 4);
  float*    dis   = (float*)alloc((size_t)NN * 4);
  float*    hwp   = (float*)alloc((size_t)NN * 4);
  float*    aggs  = (float*)alloc((size_t)NN * 4);
  float*    xsum  = (float*)alloc((size_t)BB * 156 * 4);
  _Float16* xsumh = (_Float16*)alloc((size_t)BB * 160 * 2);
  // fragment-packed weights
  _Float16* Wbp[3];
  for (int j = 0; j < 3; ++j) Wbp[j] = (_Float16*)alloc((size_t)5 * 3 * 512 * 2);
  _Float16* fcg1p = (_Float16*)alloc((size_t)8 * 5 * 512 * 2);
  _Float16* fxtp  = (_Float16*)alloc((size_t)8 * 128 * 512 * 2);
  _Float16* fc1p  = (_Float16*)alloc((size_t)64 * 8 * 512 * 2);
  _Float16* fc2p  = (_Float16*)alloc((size_t)32 * 32 * 512 * 2);
  _Float16* c1p   = (_Float16*)alloc((size_t)8 * 94 * 512 * 2);
  _Float16* c2p   = (_Float16*)alloc((size_t)4 * 12 * 512 * 2);
  _Float16* c3p   = (_Float16*)alloc((size_t)2 * 6 * 512 * 2);
  _Float16* embh  = (_Float16*)alloc((size_t)NVOC * EMBD * 2);
  // activations
  _Float16* p1h  = (_Float16*)alloc((size_t)BB * 128 * 128 * 2);
  _Float16* p2h  = (_Float16*)alloc((size_t)BB * 64 * 128 * 2);
  _Float16* p3h  = (_Float16*)alloc((size_t)BB * 32 * 128 * 2);
  _Float16* xch  = (_Float16*)alloc((size_t)BB * 256 * 2);
  _Float16* fc1h = (_Float16*)alloc((size_t)BB * 1024 * 2);
  _Float16* fc2h = (_Float16*)alloc((size_t)BB * 512 * 2);

  auto packB = [&](const float* s, _Float16* d, int K, int N, int Kt, int Nt) {
    const int threads = Nt * Kt * 32;
    k_pack_b<<<(threads + 255) / 256, 256, 0, stream>>>(s, d, K, N, Kt, Nt);
  };
  auto packA = [&](const float* s, _Float16* d, int M, int K, int Kt) {
    const int threads = (M / 16) * Kt * 32;
    k_pack_a<<<(threads + 255) / 256, 256, 0, stream>>>(s, d, M, K, Kt);
  };

  // one-time weight packing (f32 -> fragment-native f16)
  for (int j = 0; j < 3; ++j) packB(W[j], Wbp[j], FF, FF, 3, 5);
  packB(fcg1W, fcg1p, 156, 128, 5, 8);
  packB(fxtW, fxtp, 4096, 128, 128, 8);
  packB(fc1W, fc1p, 256, 1024, 8, 64);
  packB(fc2W, fc2p, 1024, 512, 32, 32);
  packA(c1W, c1p, 128, 3000, 94);
  packA(c2W, c2p, 64, 384, 12);
  packA(c3W, c3p, 32, 192, 6);
  k_f32_to_f16<<<(NVOC * EMBD + 255) / 256, 256, 0, stream>>>(emb, embh, NVOC * EMBD);

  // degrees + symmetric-norm coefficients
  hipMemsetAsync(dis, 0, (size_t)NN * 4, stream);
  k_deg_count<<<EE / 256, 256, 0, stream>>>(dst, dis);
  k_deg_finish<<<NN / 256, 256, 0, stream>>>(dis);

  // block-1 input: x -> padded f16
  k_cvt_pad<<<4096, 256, 0, stream>>>(x, hf16, NN, FF, HPAD);

  // three GCN+SAGPool blocks
  for (int j = 0; j < 3; ++j) {
    k_wmma_gemm<0, 4><<<dim3(NN / 64, 1), 160, 0, stream>>>(
        hf16, HPAD, Wbp[j], nullptr, nullptr, hw, nullptr, NN, FF, 3, 5, FF, 0);
    hipMemsetAsync(agg, 0, (size_t)NN * FF * 4, stream);
    k_edge_scatter<<<(EE * 128) / 256, 256, 0, stream>>>(hw, src, dst, dis, agg);
    k_combine<<<(NN * 128) / 256, 256, 0, stream>>>(agg, hw, dis, bv[j], bnv[j], h);
    k_pool_dot<<<NN / 256, 256, 0, stream>>>(h, Wp[j], hwp);
    hipMemsetAsync(aggs, 0, (size_t)NN * 4, stream);
    k_edge_scalar<<<EE / 256, 256, 0, stream>>>(hwp, src, dst, dis, aggs);
    k_finalize<<<(NN * 128) / 256, 256, 0, stream>>>(h, hf16, aggs, hwp, dis, bp[j]);
    k_readout<<<BB, 96, 0, stream>>>(h, xsum, j == 0);
  }

  // g = relu(bn4(xsum @ fcg1_W + b)) -> xch[:, 0:128]
  k_cvt_pad<<<(BB * 160 + 255) / 256, 256, 0, stream>>>(xsum, xsumh, BB, 156, 160);
  k_wmma_gemm<1, 2><<<dim3(BB / 32, 1), 256, 0, stream>>>(
      xsumh, 160, fcg1p, fcg1b, bn4, nullptr, xch, BB, 128, 5, 8, 256, 0);

  // protein branch: 3 convs as implicit-im2col WMMA GEMMs
  k_wmma_conv<true, 0, 8><<<BB, 256, 0, stream>>>(
      c1p, nullptr, embh, target, c1b, bnx1, p1h, 128, 94, 3000);
  k_wmma_conv<false, 128, 4><<<BB, 256, 0, stream>>>(
      c2p, p1h, nullptr, nullptr, c2b, bnx2, p2h, 64, 12, 384);
  k_wmma_conv<false, 64, 2><<<BB, 256, 0, stream>>>(
      c3p, p2h, nullptr, nullptr, c3b, bnx3, p3h, 32, 6, 192);

  // xt = bn6(relu(p3.flat @ fxt_W + b)) -> xch[:, 128:256]
  k_wmma_gemm<2, 2><<<dim3(BB / 32, 1), 256, 0, stream>>>(
      p3h, 4096, fxtp, fxtb, bn6, nullptr, xch, BB, 128, 128, 8, 256, 128);

  // FC head
  k_wmma_gemm<2, 2><<<dim3(BB / 32, 8), 256, 0, stream>>>(
      xch, 256, fc1p, fc1b, bn7, nullptr, fc1h, BB, 1024, 8, 64, 1024, 0);
  k_wmma_gemm<2, 2><<<dim3(BB / 32, 4), 256, 0, stream>>>(
      fc1h, 1024, fc2p, fc2b, bn8, nullptr, fc2h, BB, 512, 32, 32, 512, 0);
  k_out_dot<<<2, 256, 0, stream>>>(fc2h, outW, outb, out);
}
